// Model_17162689315342
// MI455X (gfx1250) — matrix operand order repaired and checked by reading, not compiled, not run on hardware
//
#include <hip/hip_runtime.h>

#define SLEN 362
#define DDIM 506
#define NH   64
#define QT   23   // ceil(362/16) query tiles

typedef __bf16 bf16_t;
typedef __attribute__((ext_vector_type(16))) __bf16 v16bf;
typedef __attribute__((ext_vector_type(8)))  __bf16 v8bf;
typedef __attribute__((ext_vector_type(8)))  float  v8f;

#define Q_STRIDE  520   // bf16 elements per Q-LDS row (16B-aligned rows)
#define SC_STRIDE 372   // f32 elements per score row
#define P_STRIDE  392   // bf16 elements per P row (16B-aligned rows)
#define VT_STRIDE 40    // bf16 elements per Vt row (80B: 16B-aligned chunks)

#define SMEM_SC_OFF 16640   // after Qlds (16*520*2)
#define SMEM_PB_OFF 40960   // after union{Qlds+Sc, Vt}
#define SMEM_BYTES  53504   // + Pb (16*392*2)

__device__ __forceinline__ unsigned int drophash(unsigned int h, unsigned int q, unsigned int k) {
  unsigned int x = (q * (unsigned)SLEN + k) * 0x9E3779B1u + h * 0x85EBCA77u;
  x ^= x >> 16; x *= 0x7feb352du;
  x ^= x >> 15; x *= 0x846ca68bu;
  x ^= x >> 16;
  return x;
}

// A fragment (16x32 bf16, M x K) from LDS row-major bf16 tile.
// K order used here (same as B frags): lane<16 -> M=lane, K in {k0..k0+15}
//               lane>=16 -> M=lane-16, K in {k0+16..k0+31}
__device__ __forceinline__ v16bf frag_a_lds(const bf16_t* base, int rowStride, int k0, int lane) {
  const int half = lane >> 4;
  const int m    = lane & 15;
  const bf16_t* row = base + m * rowStride + k0 + half * 16;
  v8bf lo = *(const v8bf*)(row);
  v8bf hi = *(const v8bf*)(row + 8);
  v16bf f;
#pragma unroll
  for (int i = 0; i < 8; ++i) { f[i] = lo[i]; f[i + 8] = hi[i]; }
  return f;
}

// B fragment (32x16 bf16, K x N) from transposed-V LDS: lane holds column
// n = nt*16 + (lane&15), contiguous K run = half*16 .. half*16+15.
__device__ __forceinline__ v16bf frag_b_vt(const bf16_t* vt, int nt, int lane) {
  const int half = lane >> 4;
  const int n    = nt * 16 + (lane & 15);
  const bf16_t* p = vt + n * VT_STRIDE + half * 16;
  v8bf lo = *(const v8bf*)(p);
  v8bf hi = *(const v8bf*)(p + 8);
  v16bf f;
#pragma unroll
  for (int i = 0; i < 8; ++i) { f[i] = lo[i]; f[i + 8] = hi[i]; }
  return f;
}

// B fragment for Q*K^T: column n of B == row n of K (row-major fp32 in global).
// Lane needs 16 contiguous d-values starting at d0 + half*16; convert to bf16.
// Row pointer is pre-clamped by the caller (scores for clamped rows are never
// consumed), so no exec-mask divergence here. Only the statically-last chunk
// (d0=480, half=1) needs the element-wise d<DDIM tail, which lowers to cndmask.
__device__ __forceinline__ v16bf frag_b_krow(const float* krow, int d0, int lane) {
  const int half = lane >> 4;
  const int c    = d0 + half * 16;
  v16bf f;
  if (d0 + 31 < DDIM) {                        // statically true for dt < 15
#pragma unroll
    for (int i = 0; i < 8; ++i) {              // 8-byte aligned (row = n*506 floats)
      float2 t = *(const float2*)(krow + c + 2 * i);
      f[2 * i]     = (bf16_t)t.x;
      f[2 * i + 1] = (bf16_t)t.y;
    }
  } else {
#pragma unroll
    for (int i = 0; i < 16; ++i) {
      int d = c + i;
      float v = krow[(d < DDIM) ? d : (DDIM - 1)];
      f[i] = (bf16_t)((d < DDIM) ? v : 0.0f);
    }
  }
  return f;
}

__global__ __launch_bounds__(256, 1)
void fattn_kernel(const float* __restrict__ Q, const float* __restrict__ K,
                  const float* __restrict__ V, const float* __restrict__ Msk,
                  float* __restrict__ O) {
  __shared__ __align__(16) unsigned char smem[SMEM_BYTES];
  bf16_t* Qlds = (bf16_t*)smem;                  // phase 1
  float*  Sc   = (float*)(smem + SMEM_SC_OFF);   // phase 1 + softmax
  bf16_t* Vt   = (bf16_t*)smem;                  // phase 2 (aliases Qlds+Sc)
  bf16_t* Pb   = (bf16_t*)(smem + SMEM_PB_OFF);  // softmax output

  const int tid  = threadIdx.x;
  const int w    = tid >> 5;   // wave id 0..7
  const int lane = tid & 31;
  const int half = lane >> 4;
  const int nl   = lane & 15;

  const int h     = blockIdx.x / QT;
  const int qt    = blockIdx.x % QT;
  const int qbase = qt * 16;

  const float* Qh = Q + (size_t)h * SLEN * DDIM;
  const float* Kh = K + (size_t)h * SLEN * DDIM;
  const float* Vh = V + (size_t)h * SLEN * DDIM;
  float*       Oh = O + (size_t)h * SLEN * DDIM;

  // ---- stage Q tile -> LDS bf16, zero-padded to 16x512 ----
  for (int f = tid; f < 16 * 512; f += 256) {
    int r = f >> 9, d = f & 511;
    int qr = qbase + r;
    float v = (qr < SLEN && d < DDIM) ? Qh[(size_t)qr * DDIM + d] : 0.0f;
    Qlds[r * Q_STRIDE + d] = (bf16_t)v;
  }
  __syncthreads();

  const float scale = 0.044455430f;  // 1/sqrt(506)

  // ---- phase 1: scores = scale * Q K^T + mask ----
  for (int kt = w; kt < QT; kt += 8) {
    const int kbase = kt * 16;
    const int n     = kbase + nl;
    // Clamp invalid key rows to the last row: their score columns (>= SLEN)
    // are never stored to Sc nor read by softmax, so no zeroing is needed.
    const int nc    = (n < SLEN) ? n : (SLEN - 1);
    const float* krow = Kh + (size_t)nc * DDIM;
    v8f c = {};
#pragma unroll
    for (int dt = 0; dt < 16; ++dt) {
      v16bf a = frag_a_lds(Qlds, Q_STRIDE, dt * 32, lane);
      v16bf b = frag_b_krow(krow, dt * 32, lane);
      c = __builtin_amdgcn_wmma_f32_16x16x32_bf16(false, a, false, b, (short)0, c, false, false);
    }
    const int col = kbase + nl;
    if (col < SLEN) {
#pragma unroll
      for (int r = 0; r < 8; ++r) {            // c[r] <-> (M = r + 8*half, N = nl)
        int m  = r + 8 * half;
        int qr = qbase + m;
        float mv = (qr < SLEN) ? Msk[(size_t)qr * SLEN + col] : 0.0f;
        Sc[m * SC_STRIDE + col] = c[r] * scale + mv;
      }
    }
  }
  __syncthreads();

  // ---- softmax + dropout -> Pb (bf16, zero-padded 16x384) ----
  {
    const int row = 2 * w + half;              // 8 waves x 2 half-waves = 16 rows
    float mx = -3.0e38f;
    for (int col = nl; col < SLEN; col += 16) mx = fmaxf(mx, Sc[row * SC_STRIDE + col]);
#pragma unroll
    for (int o = 8; o >= 1; o >>= 1) mx = fmaxf(mx, __shfl_xor(mx, o, 32));
    float sum = 0.0f;
    for (int col = nl; col < SLEN; col += 16) {
      float e = __expf(Sc[row * SC_STRIDE + col] - mx);
      Sc[row * SC_STRIDE + col] = e;
      sum += e;
    }
#pragma unroll
    for (int o = 8; o >= 1; o >>= 1) sum += __shfl_xor(sum, o, 32);
    float inv = 1.0f / sum;
    float invk = inv * 1.11111116f;            // 1/(sum * 0.9)
    for (int col = nl; col < 384; col += 16) {
      float p = 0.0f;
      if (col < SLEN) {
        unsigned int hv = drophash((unsigned)h, (unsigned)(qbase + row), (unsigned)col);
        p = (hv >= 429496730u) ? Sc[row * SC_STRIDE + col] * invk : 0.0f;
      }
      Pb[row * P_STRIDE + col] = (bf16_t)p;
    }
  }

  // ---- phase 2: O = P V, K-blocks of 32, V staged transposed in LDS ----
  v8f acc[4] = {};
  for (int kb = 0; kb < 12; ++kb) {
    __syncthreads();   // prior readers of Vt / Sc done before overwrite
    for (int f = tid; f < 32 * 512; f += 256) {
      int k = f >> 9, n = f & 511;
      int krow = kb * 32 + k;
      float v = (krow < SLEN && n < DDIM) ? Vh[(size_t)krow * DDIM + n] : 0.0f;
      Vt[n * VT_STRIDE + k] = (bf16_t)v;
    }
    __syncthreads();
    v16bf a = frag_a_lds(Pb, P_STRIDE, kb * 32, lane);
#pragma unroll
    for (int i = 0; i < 4; ++i) {
      v16bf b = frag_b_vt(Vt, w + 8 * i, lane);
      acc[i] = __builtin_amdgcn_wmma_f32_16x16x32_bf16(false, a, false, b, (short)0, acc[i], false, false);
    }
  }

  // ---- store ----
#pragma unroll
  for (int i = 0; i < 4; ++i) {
    int n = (w + 8 * i) * 16 + nl;
    if (n < DDIM) {
#pragma unroll
      for (int r = 0; r < 8; ++r) {
        int qr = qbase + r + 8 * half;
        if (qr < SLEN) Oh[(size_t)qr * DDIM + n] = acc[i][r];
      }
    }
  }
}

extern "C" void kernel_launch(void* const* d_in, const int* in_sizes, int n_in,
                              void* d_out, int out_size, void* d_ws, size_t ws_size,
                              hipStream_t stream) {
  (void)in_sizes; (void)n_in; (void)out_size; (void)d_ws; (void)ws_size;
  const float* Q = (const float*)d_in[0];
  const float* K = (const float*)d_in[1];
  const float* V = (const float*)d_in[2];
  const float* M = (const float*)d_in[3];
  float* O = (float*)d_out;
  dim3 grid(NH * QT);   // 64 heads x 23 query tiles = 1472 workgroups
  dim3 block(256);      // 8 wave32
  hipLaunchKernelGGL(fattn_kernel, grid, block, 0, stream, Q, K, V, M, O);
}
